// BandedCausalSelfAttention_48936857370739
// MI455X (gfx1250) — matrix operand order repaired and checked by reading, not compiled, not run on hardware
//
#include <hip/hip_runtime.h>

// ---------------------------------------------------------------------------
// Banded causal self-attention, MI455X (gfx1250, wave32, WMMA bf16).
//   B=1, T=4096, C=1024, H=16, DH=64, BAND=128
// Pipeline: f32->bf16 convert  ->  fused QKV GEMM (V stored transposed)
//           -> banded attention (WMMA QK^T, softmax, WMMA PV)
//           -> output projection GEMM (f32 out).
// GEMM: one wave = 64x64 tile (16 f32 accumulators, 16 B/lane/WMMA operand
// traffic); 8 waves/block as 2(M) x 4(N) over a 128x256 macro-tile so waves
// share A rows / B columns through the WGP cache.
// Workspace (~48.1 MiB): xb@0 (8M), wq/wk/wv/wo@8/10/12/14M (2M each),
// qb@16M (8M), kb@24M (8M), vT@32M (1024*4128*2), yb after vT (8M).
// ---------------------------------------------------------------------------

#define T_SEQ 4096
#define CDIM  1024
#define NH    16
#define DH    64
#define BAND  128
#define LDV   4128   // T + 32 padding so banded-edge V reads never go OOB

typedef __attribute__((ext_vector_type(16))) __bf16 v16bf;
typedef __attribute__((ext_vector_type(8)))  __bf16 v8bf;
typedef __attribute__((ext_vector_type(8)))  float  v8f;

#define WMMA_BF16(a, b, c) \
  __builtin_amdgcn_wmma_f32_16x16x32_bf16(false, (a), false, (b), (short)0, (c), false, false)

static __device__ __forceinline__ v8f zero8() {
  v8f z = {0.f, 0.f, 0.f, 0.f, 0.f, 0.f, 0.f, 0.f};
  return z;
}

// A fragment (16x32 bf16, M x K). Lanes 0-15: row M=lane, K=0..7 & 16..23;
// lanes 16-31: same rows, K=8..15 & 24..31. Source row-major.
static __device__ __forceinline__ v16bf load_a_frag(const __bf16* __restrict__ p0,
                                                    int ld, int row0, int k0, int lane) {
  const int m = lane & 15, half = lane >> 4;
  const __bf16* p = p0 + (size_t)(row0 + m) * ld + k0 + half * 8;
  union { v16bf v; v8bf h[2]; } u;
  u.h[0] = *(const v8bf*)p;          // K = k0 + half*8 + [0..7]
  u.h[1] = *(const v8bf*)(p + 16);   // K = k0 + 16 + half*8 + [0..7]
  return u.v;
}

// B fragment (32x16 bf16, K x N) fed from a source holding B^T row-major:
// lane holds column n = lane&15, K = half*8 + [0..7] & 16 + half*8 + [0..7].
static __device__ __forceinline__ v16bf load_bt_frag(const __bf16* __restrict__ p0,
                                                     int ld, int n0, int k0, int lane) {
  const int n = lane & 15, half = lane >> 4;
  const __bf16* p = p0 + (size_t)(n0 + n) * ld + k0 + half * 8;
  union { v16bf v; v8bf h[2]; } u;
  u.h[0] = *(const v8bf*)p;
  u.h[1] = *(const v8bf*)(p + 16);
  return u.v;
}

// ---------------------------------------------------------------------------
__global__ void cvt_f32_bf16(const float* __restrict__ src, __bf16* __restrict__ dst, int n) {
  int i = (blockIdx.x * 256 + threadIdx.x) * 4;
  if (i < n) {
    float4 f = *(const float4*)(src + i);
    dst[i + 0] = (__bf16)f.x;
    dst[i + 1] = (__bf16)f.y;
    dst[i + 2] = (__bf16)f.z;
    dst[i + 3] = (__bf16)f.w;
  }
}

// ---------------------------------------------------------------------------
// 64x64-per-wave GEMM core: acc += A[row0..row0+63, :] * B[:, col0..col0+63]
// A row-major (ld CDIM), W holds B^T row-major (ld CDIM).
static __device__ __forceinline__ void gemm_core_64x64(
    const __bf16* __restrict__ A, const __bf16* __restrict__ W,
    int row0, int col0, int lane, v8f acc[4][4]) {
#pragma unroll 2
  for (int k0 = 0; k0 < CDIM; k0 += 32) {
    v16bf a[4], b[4];
#pragma unroll
    for (int i = 0; i < 4; i++) a[i] = load_a_frag(A, CDIM, row0 + i * 16, k0, lane);
#pragma unroll
    for (int j = 0; j < 4; j++) b[j] = load_bt_frag(W, CDIM, col0 + j * 16, k0, lane);
#pragma unroll
    for (int i = 0; i < 4; i++)
#pragma unroll
      for (int j = 0; j < 4; j++)
        acc[i][j] = WMMA_BF16(a[i], b[j], acc[i][j]);
  }
}

// Block geometry: 8 waves = 2(M) x 4(N) over a 128x256 macro-tile.
static __device__ __forceinline__ void gemm_tile_origin(int* row0, int* col0) {
  const int w  = threadIdx.x >> 5;
  const int bn = blockIdx.x & 3;          // N/256 = 4 macro-cols
  const int bm = blockIdx.x >> 2;         // M/128 macro-rows
  *row0 = bm * 128 + (w >> 2) * 64;
  *col0 = bn * 256 + (w & 3) * 64;
}

// ---------------------------------------------------------------------------
// Fused QKV projection. blockIdx.y selects Q/K/V. Q,K stored (T,C) row-major;
// V stored transposed (C, LDV) for contiguous PV B-fragment reads.
__global__ __launch_bounds__(256) void qkv_gemm(
    const __bf16* __restrict__ xb,
    const __bf16* __restrict__ wq, const __bf16* __restrict__ wk, const __bf16* __restrict__ wv,
    __bf16* __restrict__ qb, __bf16* __restrict__ kb, __bf16* __restrict__ vT) {
  const int lane = threadIdx.x & 31;
  int row0, col0;
  gemm_tile_origin(&row0, &col0);
  const int which = blockIdx.y;
  const __bf16* __restrict__ W = (which == 0) ? wq : (which == 1) ? wk : wv;

  v8f acc[4][4];
#pragma unroll
  for (int i = 0; i < 4; i++)
#pragma unroll
    for (int j = 0; j < 4; j++) acc[i][j] = zero8();

  gemm_core_64x64(xb, W, row0, col0, lane, acc);

  const int nn = lane & 15, half = lane >> 4;
  if (which < 2) {
    __bf16* __restrict__ dst = (which == 0) ? qb : kb;
#pragma unroll
    for (int it = 0; it < 4; it++)
#pragma unroll
      for (int jt = 0; jt < 4; jt++) {
        const int col = col0 + jt * 16 + nn;
#pragma unroll
        for (int r = 0; r < 8; r++) {
          const int row = row0 + it * 16 + half * 8 + r;
          dst[(size_t)row * CDIM + col] = (__bf16)acc[it][jt][r];
        }
      }
  } else {
    // transposed store: 8 consecutive rows -> 8 contiguous bf16 (16B store)
#pragma unroll
    for (int it = 0; it < 4; it++)
#pragma unroll
      for (int jt = 0; jt < 4; jt++) {
        const int col = col0 + jt * 16 + nn;
        const int row = row0 + it * 16 + half * 8;
        v8bf tmp;
#pragma unroll
        for (int r = 0; r < 8; r++) tmp[r] = (__bf16)acc[it][jt][r];
        *(v8bf*)(vT + (size_t)col * LDV + row) = tmp;
      }
  }
}

// ---------------------------------------------------------------------------
// Banded attention: one wave per (head, 16-query tile). 9 key tiles cover the
// band; fully-masked tiles (t0 < 128) fall out via the mask -> zero weights.
__global__ __launch_bounds__(256) void attn_kernel(
    const __bf16* __restrict__ qb, const __bf16* __restrict__ kb,
    const __bf16* __restrict__ vT, __bf16* __restrict__ yb) {
  __shared__ __bf16 plds[8][16][40];   // per-wave 16x32 P staging (+8 pad)

  const int lane = threadIdx.x & 31;
  const int wave = threadIdx.x >> 5;
  const int widx = blockIdx.x * 8 + wave;      // 4096 waves = 16 heads * 256 tiles
  const int h    = widx >> 8;
  const int t0   = (widx & 255) * 16;
  const int nn = lane & 15, half = lane >> 4;
  const int jstart = (t0 >= BAND) ? (t0 - BAND) : 0;

  // Q fragments (reused across all 9 key tiles)
  v16bf qa0 = load_a_frag(qb, CDIM, t0, h * DH,      lane);
  v16bf qa1 = load_a_frag(qb, CDIM, t0, h * DH + 32, lane);

  // --- scores: S = (Q K^T) * 1/8, banded-causal mask ---
  v8f s[9];
#pragma unroll
  for (int jt = 0; jt < 9; jt++) {
    const int j0 = jstart + jt * 16;
    v16bf kf0 = load_bt_frag(kb, CDIM, j0, h * DH,      lane);
    v16bf kf1 = load_bt_frag(kb, CDIM, j0, h * DH + 32, lane);
    v8f acc = zero8();
    acc = WMMA_BF16(qa0, kf0, acc);
    acc = WMMA_BF16(qa1, kf1, acc);
    const int j = j0 + nn;
#pragma unroll
    for (int r = 0; r < 8; r++) {
      const int i = t0 + half * 8 + r;
      const bool valid = (j <= i) && ((i - j) < BAND);
      acc[r] = valid ? acc[r] * 0.125f : -1.0e30f;
    }
    s[jt] = acc;
  }

  // --- softmax: rows live across the 16 lanes of each half-wave ---
  float rmax[8], rsum[8];
#pragma unroll
  for (int r = 0; r < 8; r++) {
    float m = s[0][r];
#pragma unroll
    for (int jt = 1; jt < 9; jt++) m = fmaxf(m, s[jt][r]);
#pragma unroll
    for (int off = 1; off < 16; off <<= 1) m = fmaxf(m, __shfl_xor(m, off, 32));
    rmax[r] = m;
  }
#pragma unroll
  for (int r = 0; r < 8; r++) {
    float acc = 0.f;
#pragma unroll
    for (int jt = 0; jt < 9; jt++) {
      float p = __expf(s[jt][r] - rmax[r]);
      s[jt][r] = p;
      acc += p;
    }
#pragma unroll
    for (int off = 1; off < 16; off <<= 1) acc += __shfl_xor(acc, off, 32);
    rsum[r] = acc;
  }

  // --- PV: restage P (C/D layout) -> A layout through LDS, 5 K-chunks of 32 ---
  v8f yacc[4] = {zero8(), zero8(), zero8(), zero8()};
#pragma unroll
  for (int c = 0; c < 5; c++) {
#pragma unroll
    for (int sub = 0; sub < 2; sub++) {
      const int jt = c * 2 + sub;
#pragma unroll
      for (int r = 0; r < 8; r++) {
        const float p = (jt < 9) ? s[jt][r] : 0.f;
        plds[wave][half * 8 + r][sub * 16 + nn] = (__bf16)p;   // lane = column nn
      }
    }
    __builtin_amdgcn_wave_barrier();   // keep DS store->load order (same-wave, in-order HW)
    union { v16bf v; v8bf hh[2]; } ua;
    const __bf16* pl = &plds[wave][nn][0];                     // lane = row nn (A layout)
    ua.hh[0] = *(const v8bf*)(pl + half * 8);
    ua.hh[1] = *(const v8bf*)(pl + 16 + half * 8);

    const int jb = jstart + c * 32;   // >= 0; reads bounded by LDV padding
#pragma unroll
    for (int dt = 0; dt < 4; dt++) {
      v16bf vf = load_bt_frag(vT, LDV, h * DH + dt * 16, jb, lane);
      yacc[dt] = WMMA_BF16(ua.v, vf, yacc[dt]);
    }
    __builtin_amdgcn_wave_barrier();
  }

  // --- normalize and store Y (T, C) bf16 ---
#pragma unroll
  for (int r = 0; r < 8; r++) {
    const float inv = 1.0f / rsum[r];
    const int row = t0 + half * 8 + r;
#pragma unroll
    for (int dt = 0; dt < 4; dt++) {
      const int col = h * DH + dt * 16 + nn;
      yb[(size_t)row * CDIM + col] = (__bf16)(yacc[dt][r] * inv);
    }
  }
}

// ---------------------------------------------------------------------------
// Output projection: out = Y @ Wo^T, f32 result straight to d_out.
__global__ __launch_bounds__(256) void out_proj_gemm(
    const __bf16* __restrict__ yb, const __bf16* __restrict__ wo,
    float* __restrict__ out) {
  const int lane = threadIdx.x & 31;
  int row0, col0;
  gemm_tile_origin(&row0, &col0);

  v8f acc[4][4];
#pragma unroll
  for (int i = 0; i < 4; i++)
#pragma unroll
    for (int j = 0; j < 4; j++) acc[i][j] = zero8();

  gemm_core_64x64(yb, wo, row0, col0, lane, acc);

  const int nn = lane & 15, half = lane >> 4;
#pragma unroll
  for (int it = 0; it < 4; it++)
#pragma unroll
    for (int jt = 0; jt < 4; jt++) {
      const int col = col0 + jt * 16 + nn;
#pragma unroll
      for (int r = 0; r < 8; r++) {
        const int row = row0 + it * 16 + half * 8 + r;
        out[(size_t)row * CDIM + col] = acc[it][jt][r];
      }
    }
}

// ---------------------------------------------------------------------------
extern "C" void kernel_launch(void* const* d_in, const int* in_sizes, int n_in,
                              void* d_out, int out_size, void* d_ws, size_t ws_size,
                              hipStream_t stream) {
  (void)in_sizes; (void)n_in; (void)out_size; (void)ws_size;

  const float* x  = (const float*)d_in[0];
  const float* Wq = (const float*)d_in[1];
  const float* Wk = (const float*)d_in[2];
  const float* Wv = (const float*)d_in[3];
  const float* Wo = (const float*)d_in[4];

  char* ws = (char*)d_ws;
  const size_t MB = 1024u * 1024u;
  __bf16* xb  = (__bf16*)(ws);
  __bf16* wqb = (__bf16*)(ws + 8 * MB);
  __bf16* wkb = (__bf16*)(ws + 10 * MB);
  __bf16* wvb = (__bf16*)(ws + 12 * MB);
  __bf16* wob = (__bf16*)(ws + 14 * MB);
  __bf16* qb  = (__bf16*)(ws + 16 * MB);
  __bf16* kb  = (__bf16*)(ws + 24 * MB);
  __bf16* vT  = (__bf16*)(ws + 32 * MB);                       // 1024*4128*2 bytes
  __bf16* yb  = (__bf16*)(ws + 32 * MB + (size_t)CDIM * LDV * 2);

  // 1) f32 -> bf16 conversions (4 elems/thread)
  cvt_f32_bf16<<<(T_SEQ * CDIM) / 1024, 256, 0, stream>>>(x, xb, T_SEQ * CDIM);
  cvt_f32_bf16<<<(CDIM * CDIM) / 1024, 256, 0, stream>>>(Wq, wqb, CDIM * CDIM);
  cvt_f32_bf16<<<(CDIM * CDIM) / 1024, 256, 0, stream>>>(Wk, wkb, CDIM * CDIM);
  cvt_f32_bf16<<<(CDIM * CDIM) / 1024, 256, 0, stream>>>(Wv, wvb, CDIM * CDIM);
  cvt_f32_bf16<<<(CDIM * CDIM) / 1024, 256, 0, stream>>>(Wo, wob, CDIM * CDIM);

  // 2) fused QKV projection: (T/128)*(C/256)=128 blocks per matrix, x3
  qkv_gemm<<<dim3(128, 3, 1), 256, 0, stream>>>(xb, wqb, wkb, wvb, qb, kb, vT);

  // 3) banded attention: 16 heads * 256 query tiles = 4096 waves
  attn_kernel<<<512, 256, 0, stream>>>(qb, kb, vT, yb);

  // 4) output projection, f32 result
  out_proj_gemm<<<128, 256, 0, stream>>>(yb, wob, (float*)d_out);
}
